// TorchGroupGemmReduceRS_31997506355742
// MI455X (gfx1250) — matrix operand order repaired and checked by reading, not compiled, not run on hardware
//
#include <hip/hip_runtime.h>
#include <hip/hip_bf16.h>

// ---------------- types ----------------
typedef __bf16 bf16;
typedef bf16  v2bf  __attribute__((ext_vector_type(2)));
typedef bf16  v8bf  __attribute__((ext_vector_type(8)));
typedef bf16  v16bf __attribute__((ext_vector_type(16)));
typedef float v8f   __attribute__((ext_vector_type(8)));

union VFrag { v16bf v; v8bf h[2]; };

// ---------------- problem constants ----------------
#define T_ROWS   16384
#define I_DIM    4096
#define H_DIM    4096
#define N_EXP    8
#define BM       128
#define BN       128
#define BK       32
#define LDSK     56              // bf16 row stride: 112B -> 16B-aligned b128, conflict-free across 16 lanes
#define MAX_TILES 136            // 16384/128 + 8 experts of padding
#define PERM_CAP  (MAX_TILES * BM)

__device__ __forceinline__ void splitbf(float x, bf16& h, bf16& l) {
    h = (bf16)x;
    l = (bf16)(x - (float)h);
}

// ---------------- kernel 0: zero output ----------------
__global__ void zero_out_kernel(float4* __restrict__ out, int n4) {
    int i = blockIdx.x * blockDim.x + threadIdx.x;
    if (i < n4) out[i] = make_float4(0.f, 0.f, 0.f, 0.f);
}

// ---------------- kernel 1: deterministic routing ----------------
__global__ __launch_bounds__(256) void route_kernel(const int* __restrict__ ids,
                                                    int* __restrict__ perm,
                                                    int* __restrict__ tileExpert,
                                                    int* __restrict__ tileBase) {
    __shared__ int cnt[256][N_EXP];
    __shared__ int expOff[N_EXP + 1];
    __shared__ int expTotal[N_EXP];

    const int t = threadIdx.x;
    const int ROWS_PER_THREAD = T_ROWS / 256;   // 64

    int lc[N_EXP];
#pragma unroll
    for (int e = 0; e < N_EXP; ++e) lc[e] = 0;
    for (int r = 0; r < ROWS_PER_THREAD; ++r) {
        int e = ids[t * ROWS_PER_THREAD + r] & 7;
        lc[e]++;
    }
#pragma unroll
    for (int e = 0; e < N_EXP; ++e) cnt[t][e] = lc[e];
    __syncthreads();

    if (t < N_EXP) {
        int run = 0;
        for (int i = 0; i < 256; ++i) {
            int c = cnt[i][t];
            cnt[i][t] = run;
            run += c;
        }
        expTotal[t] = run;
    }
    __syncthreads();

    if (t == 0) {
        int off = 0, tile = 0;
        for (int e = 0; e < N_EXP; ++e) {
            expOff[e] = off;
            int padded = (expTotal[e] + BM - 1) / BM * BM;
            int nt = padded / BM;
            for (int i = 0; i < nt; ++i) {
                tileExpert[tile] = e;
                tileBase[tile]   = off + i * BM;
                ++tile;
            }
            off += padded;
        }
        expOff[N_EXP] = off;
        for (int q = tile; q < MAX_TILES; ++q) { tileExpert[q] = -1; tileBase[q] = 0; }
    }
    __syncthreads();

    const int total = expOff[N_EXP];
    for (int i = t; i < total; i += 256) perm[i] = -1;
    __threadfence();
    __syncthreads();

#pragma unroll
    for (int e = 0; e < N_EXP; ++e) lc[e] = cnt[t][e];
    for (int r = 0; r < ROWS_PER_THREAD; ++r) {
        int row = t * ROWS_PER_THREAD + r;
        int e = ids[row] & 7;
        perm[expOff[e] + lc[e]++] = row;
    }
}

// ---------------- kernel 2: grouped GEMM via bf16x3 WMMA ----------------
// Block: 256 threads (8 waves). Block tile 128M x 128N, BK=32.
// Wave tile: 32M x 64N = 8 accumulators -> 24 WMMA per K-step per wave.
__global__ __launch_bounds__(256) void moe_gemm_kernel(const float* __restrict__ X,
                                                       const float* __restrict__ W,
                                                       const float* __restrict__ wts,
                                                       const int* __restrict__ perm,
                                                       const int* __restrict__ tileExpert,
                                                       const int* __restrict__ tileBase,
                                                       float* __restrict__ out) {
    const int mt = blockIdx.y;
    const int e  = tileExpert[mt];
    if (e < 0) return;

    __shared__ alignas(16) bf16 sAhi[BM * LDSK];
    __shared__ alignas(16) bf16 sAlo[BM * LDSK];
    __shared__ alignas(16) bf16 sBhi[BN * LDSK];   // transposed: [col][k]
    __shared__ alignas(16) bf16 sBlo[BN * LDSK];
    __shared__ int   sRow[BM];
    __shared__ float sW[BM];

    const int tid  = threadIdx.x;
    const int lane = tid & 31;
    const int wid  = tid >> 5;        // 8 waves
    const int wm   = wid & 3;         // 4 row groups of 32
    const int wn   = wid >> 2;        // 2 col groups of 64

    const int rbase = tileBase[mt];
    if (tid < 128) {
        int row = perm[rbase + tid];
        sRow[tid] = row;
        sW[tid]   = (row >= 0) ? wts[row] : 0.f;
    }
    __syncthreads();

    const float* Wexp  = W + (size_t)e * I_DIM * H_DIM;
    const int    nBase = blockIdx.x * BN;

    v8f acc[2][4] = {};

    // ---- staging coordinates ----
    // A: 128 rows x 32 K, 16 floats/thread (one row-half of K)
    const int aRow = tid >> 1;            // 0..127
    const int aSeg = (tid & 1) * 16;      // 0 or 16
    // padded rows (-1): clamp to row 0; garbage stays in that M lane and the
    // epilogue skips it, so no zero-fill branch is needed.
    const int aRowId = sRow[aRow];
    const float* aPtr = X + (size_t)(aRowId < 0 ? 0 : aRowId) * I_DIM + aSeg;
    // B: 32 K x 128 N, 2K x 8N patch per thread
    const int bKp = (tid >> 4) * 2;       // 0,2,..,30
    const int bCg = (tid & 15) * 8;       // 0..120
    const float* bPtr = Wexp + (size_t)bKp * H_DIM + nBase + bCg;

    // ---- fragment coordinates (ISA 7.12.2, 16-bit layouts) ----
    const int m16  = lane & 15;
    const int ksel = lane >> 4;           // 0: K{0..7,16..23}  1: K{8..15,24..31}
    const int aoff = (wm * 32 + m16) * LDSK + ksel * 8;   // M-group 0; group 1 at +16*LDSK

    // ---- register prefetch buffers ----
    float4 pa0, pa1, pa2, pa3, pb00, pb01, pb10, pb11;

    auto load_tiles = [&](int k0) {
        const float4* p = (const float4*)(aPtr + k0);
        pa0 = p[0]; pa1 = p[1]; pa2 = p[2]; pa3 = p[3];
        const float4* q0 = (const float4*)(bPtr + (size_t)k0 * H_DIM);
        const float4* q1 = (const float4*)(bPtr + (size_t)(k0 + 1) * H_DIM);
        pb00 = q0[0]; pb01 = q0[1];
        pb10 = q1[0]; pb11 = q1[1];
    };

    auto store_tiles = [&]() {
        // A: 16 contiguous bf16 per array -> 2 x b128 stores each
        float av[16] = {pa0.x, pa0.y, pa0.z, pa0.w, pa1.x, pa1.y, pa1.z, pa1.w,
                        pa2.x, pa2.y, pa2.z, pa2.w, pa3.x, pa3.y, pa3.z, pa3.w};
        v8bf hv0, lv0, hv1, lv1;
#pragma unroll
        for (int j = 0; j < 8; ++j) {
            bf16 h, l; splitbf(av[j], h, l);
            hv0[j] = h; lv0[j] = l;
        }
#pragma unroll
        for (int j = 0; j < 8; ++j) {
            bf16 h, l; splitbf(av[8 + j], h, l);
            hv1[j] = h; lv1[j] = l;
        }
        const int abase = aRow * LDSK + aSeg;
        *(v8bf*)&sAhi[abase]     = hv0;
        *(v8bf*)&sAhi[abase + 8] = hv1;
        *(v8bf*)&sAlo[abase]     = lv0;
        *(v8bf*)&sAlo[abase + 8] = lv1;

        // B: transpose with packed v2bf (two K values per column) stores
        float b0[8] = {pb00.x, pb00.y, pb00.z, pb00.w, pb01.x, pb01.y, pb01.z, pb01.w};
        float b1[8] = {pb10.x, pb10.y, pb10.z, pb10.w, pb11.x, pb11.y, pb11.z, pb11.w};
#pragma unroll
        for (int j = 0; j < 8; ++j) {
            bf16 h0, l0, h1, l1;
            splitbf(b0[j], h0, l0);
            splitbf(b1[j], h1, l1);
            v2bf hp, lp;
            hp[0] = h0; hp[1] = h1;
            lp[0] = l0; lp[1] = l1;
            *(v2bf*)&sBhi[(bCg + j) * LDSK + bKp] = hp;
            *(v2bf*)&sBlo[(bCg + j) * LDSK + bKp] = lp;
        }
    };

    load_tiles(0);

    for (int k0 = 0; k0 < I_DIM; k0 += BK) {
        __syncthreads();            // previous LDS readers done
        store_tiles();
        __syncthreads();            // LDS tiles ready
        if (k0 + BK < I_DIM) load_tiles(k0 + BK);   // overlap next global loads with WMMAs

        VFrag aH[2], aL[2];
#pragma unroll
        for (int mg = 0; mg < 2; ++mg) {
            const int ao = aoff + mg * 16 * LDSK;
            aH[mg].h[0] = *(const v8bf*)&sAhi[ao];
            aH[mg].h[1] = *(const v8bf*)&sAhi[ao + 16];
            aL[mg].h[0] = *(const v8bf*)&sAlo[ao];
            aL[mg].h[1] = *(const v8bf*)&sAlo[ao + 16];
        }

#pragma unroll
        for (int ct = 0; ct < 4; ++ct) {
            const int boff = (wn * 64 + ct * 16 + m16) * LDSK + ksel * 8;
            VFrag bH, bL;
            bH.h[0] = *(const v8bf*)&sBhi[boff];
            bH.h[1] = *(const v8bf*)&sBhi[boff + 16];
            bL.h[0] = *(const v8bf*)&sBlo[boff];
            bL.h[1] = *(const v8bf*)&sBlo[boff + 16];

#pragma unroll
            for (int mg = 0; mg < 2; ++mg) {
                acc[mg][ct] = __builtin_amdgcn_wmma_f32_16x16x32_bf16(
                    false, aH[mg].v, false, bH.v, (short)0, acc[mg][ct], false, false);
                acc[mg][ct] = __builtin_amdgcn_wmma_f32_16x16x32_bf16(
                    false, aH[mg].v, false, bL.v, (short)0, acc[mg][ct], false, false);
                acc[mg][ct] = __builtin_amdgcn_wmma_f32_16x16x32_bf16(
                    false, aL[mg].v, false, bH.v, (short)0, acc[mg][ct], false, false);
            }
        }
    }

    // ---- epilogue: scale by routing weight, atomic-add into out[token] ----
    // C layout: VGPR r -> M = r + 8*(lane>=16), N = lane&15
    const int nloc = lane & 15;
    const int mh   = lane >> 4;
    const int colB = nBase + wn * 64;
#pragma unroll
    for (int mg = 0; mg < 2; ++mg) {
#pragma unroll
        for (int ct = 0; ct < 4; ++ct) {
#pragma unroll
            for (int r = 0; r < 8; ++r) {
                int mloc = wm * 32 + mg * 16 + mh * 8 + r;
                int row  = sRow[mloc];
                if (row >= 0) {
                    float v = acc[mg][ct][r] * sW[mloc];
                    unsafeAtomicAdd(&out[(size_t)(row >> 1) * H_DIM + colB + ct * 16 + nloc], v);
                }
            }
        }
    }
}

// ---------------- launcher ----------------
extern "C" void kernel_launch(void* const* d_in, const int* in_sizes, int n_in,
                              void* d_out, int out_size, void* d_ws, size_t ws_size,
                              hipStream_t stream) {
    const float* X   = (const float*)d_in[0];   // [T, I]
    const float* W   = (const float*)d_in[1];   // [E, I, H]
    const float* wts = (const float*)d_in[2];   // [T/2, 2] flat
    const int*   ids = (const int*)d_in[3];     // [T/2, 2] flat

    float* out = (float*)d_out;                 // [T/2, H]

    int* perm       = (int*)d_ws;               // PERM_CAP
    int* tileExpert = perm + PERM_CAP;          // MAX_TILES
    int* tileBase   = tileExpert + 144;         // MAX_TILES (padded stride)

    const int n4 = (T_ROWS / 2) * H_DIM / 4;
    zero_out_kernel<<<(n4 + 255) / 256, 256, 0, stream>>>((float4*)out, n4);

    route_kernel<<<1, 256, 0, stream>>>(ids, perm, tileExpert, tileBase);

    dim3 grid(H_DIM / BN, MAX_TILES);
    moe_gemm_kernel<<<grid, 256, 0, stream>>>(X, W, wts, perm, tileExpert, tileBase, out);
}
